// AdaAttN_new_IN_83502754169130
// MI455X (gfx1250) — matrix operand order repaired and checked by reading, not compiled, not run on hardware
//
#include <hip/hip_runtime.h>
#include <math.h>

typedef __bf16 bf16;
typedef __attribute__((ext_vector_type(16))) __bf16 v16bf;
typedef __attribute__((ext_vector_type(8)))  __bf16 v8bf;
typedef __attribute__((ext_vector_type(8)))  float  v8f;

// ---------------------------------------------------------------------------
// WMMA helpers (CDNA5 wave32, V_WMMA_F32_16X16X32_BF16)
// ---------------------------------------------------------------------------
__device__ __forceinline__ v8f wmma_bf16(v16bf a, v16bf b, v8f c) {
  return __builtin_amdgcn_wmma_f32_16x16x32_bf16(false, a, false, b, (short)0, c,
                                                 false, false);
}

// A fragment (16M x 32K) from an (m-major, k-contiguous) bf16 buffer.
// Layout: lanes 0-15 row=lane, K=0..15; lanes 16-31 row=lane-16,
// K=16..31  -> two contiguous 16B chunks per lane.
__device__ __forceinline__ v16bf load_a(const bf16* base, int ld, int lane) {
  int row = lane & 15;
  int ko  = (lane < 16) ? 0 : 16;
  const bf16* p = base + row * ld + ko;
  v8bf lo = *(const v8bf*)(p);
  v8bf hi = *(const v8bf*)(p + 8);
  v16bf r;
#pragma unroll
  for (int i = 0; i < 8; ++i) { r[i] = lo[i]; r[8 + i] = hi[i]; }
  return r;
}

// B fragment (32K x 16N) from an (n-major, k-contiguous) bf16 buffer.
// lanes 0-15: col=lane, K=0..15; lanes 16-31: col=lane-16, K=16..31.
__device__ __forceinline__ v16bf load_b(const bf16* base, int ld, int lane) {
  int n  = lane & 15;
  int ko = (lane < 16) ? 0 : 16;
  return *(const v16bf*)(base + n * ld + ko);
}

// CDNA5 async global->LDS DMA (ASYNCcnt-tracked), 16B per lane.
__device__ __forceinline__ void async_ld16(unsigned lds_off, const void* gaddr) {
  asm volatile("global_load_async_to_lds_b128 %0, %1, off"
               :: "v"(lds_off), "v"((unsigned long long)(uintptr_t)gaddr)
               : "memory");
}
// Low 32 bits of a generic shared-memory address == LDS byte offset.
__device__ __forceinline__ unsigned lds_off(const void* p) {
  return (unsigned)(uintptr_t)p;
}

// ---------------------------------------------------------------------------
// Per-channel instance-norm statistics (biased variance, eps = 1e-5)
// transposed=1: X is (L x C), channel is a column.  transposed=0: X is (C x L).
// ---------------------------------------------------------------------------
__global__ void stats_kernel(const float* __restrict__ X, float* __restrict__ mean,
                             float* __restrict__ rstd, int C, int L, int transposed) {
  __shared__ float red[256], red2[256];
  int c = blockIdx.x, t = threadIdx.x;
  float s = 0.f, s2 = 0.f;
  if (transposed) {
    for (int i = t; i < L; i += 256) { float x = X[(size_t)i * C + c]; s += x; s2 += x * x; }
  } else {
    const float* row = X + (size_t)c * L;
    for (int i = t; i < L; i += 256) { float x = row[i]; s += x; s2 += x * x; }
  }
  red[t] = s; red2[t] = s2; __syncthreads();
  for (int o = 128; o > 0; o >>= 1) {
    if (t < o) { red[t] += red[t + o]; red2[t] += red2[t + o]; }
    __syncthreads();
  }
  if (t == 0) {
    float m = red[0] / (float)L;
    float v = red2[0] / (float)L - m * m;
    if (v < 0.f) v = 0.f;
    mean[c] = m;
    rstd[c] = rsqrtf(v + 1e-5f);
  }
}

// ---------------------------------------------------------------------------
// Fold instance-norm (and the C^-0.5 softmax scale for the q path) into the
// 1x1-conv weights; emit bf16 weights + f32 folded biases.
// ---------------------------------------------------------------------------
__global__ void fold_kernel(const float* __restrict__ qw, const float* __restrict__ qb,
                            const float* __restrict__ kw, const float* __restrict__ kb,
                            const float* __restrict__ sw, const float* __restrict__ sb,
                            const float* __restrict__ mq, const float* __restrict__ rq,
                            const float* __restrict__ mk, const float* __restrict__ rk,
                            bf16* __restrict__ qwf, float* __restrict__ qbf,
                            bf16* __restrict__ kwf, float* __restrict__ kbf,
                            bf16* __restrict__ swf, float* __restrict__ sbf, int C,
                            float qscale) {
  __shared__ float red[256];
  int co = blockIdx.x, ci = threadIdx.x;
  size_t idx = (size_t)co * C + ci;
  float wq = qw[idx] * rq[ci] * qscale;   // rstd fold + softmax scale fold
  float wk = kw[idx] * rk[ci];            // rstd fold
  qwf[idx] = (bf16)wq;
  kwf[idx] = (bf16)wk;
  swf[idx] = (bf16)sw[idx];

  red[ci] = wq * mq[ci]; __syncthreads();
  for (int o = 128; o > 0; o >>= 1) { if (ci < o) red[ci] += red[ci + o]; __syncthreads(); }
  float qsum = red[0]; __syncthreads();
  red[ci] = wk * mk[ci]; __syncthreads();
  for (int o = 128; o > 0; o >>= 1) { if (ci < o) red[ci] += red[ci + o]; __syncthreads(); }
  if (ci == 0) {
    qbf[co] = qb[co] * qscale - qsum;
    kbf[co] = kb[co] - red[0];
    sbf[co] = sb[co];
  }
}

// ---------------------------------------------------------------------------
// Generic 256-out-channel 1x1 conv:  Y = W(256x256,bf16) @ X + bias
// mode 0: X = (Nn x 256) f32 row-major (n-major/k-contig), T-store Y (Nn x 256)
// mode 1: X = (256 x Nn) f32 (k-major), transpose-stage,  T-store Y (Nn x 256)
// mode 2: like 1 but direct-store Y (256 x Nn) and Y2 = Y*Y (256 x Nn)
// ---------------------------------------------------------------------------
#define LNB 64
__global__ __launch_bounds__(256) void linear_kernel(
    const bf16* __restrict__ W, const float* __restrict__ bias,
    const float* __restrict__ X, bf16* __restrict__ Y, bf16* __restrict__ Y2,
    int Nn, int mode) {
  __shared__ bf16 Xblk[LNB * 256];  // (n-major, k-contiguous) B-source
  int tid = threadIdx.x, lane = tid & 31, wave = tid >> 5;
  int n0g = blockIdx.x * LNB;
  int hi8 = (lane < 16) ? 0 : 8;

  if (mode == 0) {
    for (int i = tid; i < LNB * 256; i += 256) {
      int r = i >> 8, c = i & 255;                 // coalesced along c
      Xblk[i] = (bf16)X[(size_t)(n0g + r) * 256 + c];
    }
  } else {
    for (int i = tid; i < LNB * 256; i += 256) {
      int c = i >> 6, j = i & 63;                  // coalesced along n
      Xblk[j * 256 + c] = (bf16)X[(size_t)c * Nn + n0g + j];
    }
  }
  __syncthreads();

  for (int t = wave; t < 16 * (LNB / 16); t += 8) {
    int m0 = (t >> 2) * 16, n0 = (t & 3) * 16;
    v8f d = {};
#pragma unroll
    for (int k = 0; k < 256; k += 32) {
      v16bf a = load_a(W + (size_t)m0 * 256 + k, 256, lane);
      v16bf b = load_b(Xblk + n0 * 256 + k, 256, lane);
      d = wmma_bf16(a, b, d);
    }
#pragma unroll
    for (int v = 0; v < 8; ++v) d[v] += bias[m0 + v + hi8];

    if (mode == 2) {
#pragma unroll
      for (int v = 0; v < 8; ++v) {
        size_t off = (size_t)(m0 + v + hi8) * Nn + n0g + n0 + (lane & 15);
        float val = d[v];
        Y[off]  = (bf16)val;
        Y2[off] = (bf16)(val * val);
      }
    } else {  // transposed store: 8 contiguous bf16 per lane (one 16B store)
      v8bf o;
#pragma unroll
      for (int v = 0; v < 8; ++v) o[v] = (bf16)d[v];
      *(v8bf*)(Y + (size_t)(n0g + n0 + (lane & 15)) * 256 + m0 + hi8) = o;
    }
  }
}

// ---------------------------------------------------------------------------
// Fused flash-style attention + weighted moments + AdaAttN epilogue.
// Workgroup = 64 q-rows, 8 waves. Streams HW in 64-wide tiles with
// DOUBLE-BUFFERED async global->LDS staging (ASYNCcnt) overlapping WMMA.
// ---------------------------------------------------------------------------
#define MBLK 64
#define HWT  64

// Per-wave share of one tile stage: 8x512B of keT (contiguous) plus
// 8x(2x512B) of se/se2 (4 rows of 128B per instruction). 24 async ops/wave.
__device__ __forceinline__ void stage_tile_async(
    const bf16* __restrict__ keT, const bf16* __restrict__ se,
    const bf16* __restrict__ se2, bf16* keD, bf16* seD, bf16* s2D,
    int hw0, int HW, int wave, int lane) {
  {  // ke tile: flat 32KB copy from (HW,C) layout
    const char* g = (const char*)(keT + (size_t)hw0 * 256);
    unsigned l0 = lds_off(keD);
    int base = wave * 4096 + lane * 16;
#pragma unroll
    for (int i = 0; i < 8; ++i) {
      int off = base + i * 512;
      async_ld16(l0 + off, g + off);
    }
  }
  {  // se / se2 tiles: 256 rows x 128B from (C,HW) layout
    const char* g1 = (const char*)se  + (size_t)hw0 * 2;
    const char* g2 = (const char*)se2 + (size_t)hw0 * 2;
    unsigned l1 = lds_off(seD);
    unsigned l2 = lds_off(s2D);
    int rl = lane >> 3;            // 4 rows per instruction
    int co = (lane & 7) * 16;      // 8 lanes cover one 128B row
    int r0 = wave * 32;
#pragma unroll
    for (int i = 0; i < 8; ++i) {
      int r = r0 + i * 4 + rl;
      size_t gr = (size_t)r * HW * 2 + co;
      int    lr = r * 128 + co;
      async_ld16(l1 + lr, g1 + gr);
      async_ld16(l2 + lr, g2 + gr);
    }
  }
}

__global__ __launch_bounds__(256, 1) void flash_kernel(
    const bf16* __restrict__ qe, const bf16* __restrict__ keT,
    const bf16* __restrict__ se, const bf16* __restrict__ se2,
    const float* __restrict__ q, const float* __restrict__ mq,
    const float* __restrict__ rq, float* __restrict__ out, int HW) {
  extern __shared__ char smem[];
  bf16* qeblk  = (bf16*)smem;                    // [MBLK][256] A-source (qe)
  bf16* keb[2] = {qeblk + MBLK * 256, qeblk + MBLK * 256 + HWT * 256};
  bf16* seb[2] = {keb[1] + HWT * 256, keb[1] + 2 * HWT * 256};
  bf16* s2b[2] = {seb[1] + 256 * HWT, seb[1] + 2 * 256 * HWT};
  bf16* pblk   = s2b[1] + 256 * HWT;             // [MBLK][HWT] A-source (probs)
  float* Sp    = (float*)(pblk + MBLK * HWT);    // [2][MBLK] row sum-of-exp

  int tid = threadIdx.x, lane = tid & 31, wave = tid >> 5;
  int hi8 = (lane < 16) ? 0 : 8;
  int rb = wave >> 1, half = wave & 1;
  int n0g = blockIdx.x * MBLK;

  {  // stage qe block once (row-major, coalesced 16B copies)
    const uint4* src = (const uint4*)(qe + (size_t)n0g * 256);
    uint4* dst = (uint4*)qeblk;
    for (int i = tid; i < MBLK * 32; i += 256) dst[i] = src[i];
  }
  // prologue: async-stage tile 0 into buffer 0
  stage_tile_async(keT, se, se2, keb[0], seb[0], s2b[0], 0, HW, wave, lane);

  float sacc[8];
  v8f accM[8], accE[8];
#pragma unroll
  for (int v = 0; v < 8; ++v) sacc[v] = 0.f;
  v8f zero = {};
#pragma unroll
  for (int nt = 0; nt < 8; ++nt) { accM[nt] = zero; accE[nt] = zero; }

  const int nsteps = HW / HWT;
  for (int s = 0; s < nsteps; ++s) {
    int hw0 = s * HWT;
    int cur = s & 1;
    __syncthreads();  // prev iteration fully consumed buf[cur^1] and pblk

    if (s + 1 < nsteps) {
      // prefetch next tile into the other buffer, then wait for current tile:
      // async loads complete in order, so <=24 outstanding => tile s landed.
      stage_tile_async(keT, se, se2, keb[cur ^ 1], seb[cur ^ 1], s2b[cur ^ 1],
                       hw0 + HWT, HW, wave, lane);
      asm volatile("s_wait_asynccnt 0x18" ::: "memory");
    } else {
      asm volatile("s_wait_asynccnt 0x0" ::: "memory");
    }
    __syncthreads();  // tile s visible to all waves

    // ---- Phase A: logits -> exp -> row-sum, write probs to LDS ----
    {
      const bf16* ke = keb[cur];
      int m0 = rb * 16, n0 = half * 32;
      v8f d0 = {}, d1 = {};
#pragma unroll
      for (int k = 0; k < 256; k += 32) {
        v16bf a  = load_a(qeblk + (size_t)m0 * 256 + k, 256, lane);
        v16bf b0 = load_b(ke + (size_t)n0 * 256 + k, 256, lane);
        v16bf b1 = load_b(ke + (size_t)(n0 + 16) * 256 + k, 256, lane);
        d0 = wmma_bf16(a, b0, d0);
        d1 = wmma_bf16(a, b1, d1);
      }
#pragma unroll
      for (int v = 0; v < 8; ++v) {
        float e0 = __expf(d0[v]);
        float e1 = __expf(d1[v]);
        sacc[v] += e0 + e1;
        int r = m0 + v + hi8;
        pblk[r * HWT + n0 + (lane & 15)]      = (bf16)e0;
        pblk[r * HWT + n0 + 16 + (lane & 15)] = (bf16)e1;
      }
    }
    __syncthreads();

    // ---- Phase B: acc += p @ se, p @ se^2 (rows rb*16.., cols half*128..) ----
    {
      const bf16* sse = seb[cur];
      const bf16* ss2 = s2b[cur];
      int m0 = rb * 16;
#pragma unroll
      for (int kk = 0; kk < HWT; kk += 32) {
        v16bf a = load_a(pblk + (size_t)m0 * HWT + kk, HWT, lane);
#pragma unroll
        for (int nt = 0; nt < 8; ++nt) {
          int c0 = half * 128 + nt * 16;
          v16bf b  = load_b(sse + (size_t)c0 * HWT + kk, HWT, lane);
          v16bf b2 = load_b(ss2 + (size_t)c0 * HWT + kk, HWT, lane);
          accM[nt] = wmma_bf16(a, b,  accM[nt]);
          accE[nt] = wmma_bf16(a, b2, accE[nt]);
        }
      }
    }
  }

  // ---- row sum-of-exp: reduce across 16-lane halves, combine hw halves ----
#pragma unroll
  for (int v = 0; v < 8; ++v) {
    float s = sacc[v];
    s += __shfl_xor(s, 1, 32);
    s += __shfl_xor(s, 2, 32);
    s += __shfl_xor(s, 4, 32);
    s += __shfl_xor(s, 8, 32);
    sacc[v] = s;
  }
  if (lane == 0)
    for (int v = 0; v < 8; ++v) Sp[half * MBLK + rb * 16 + v] = sacc[v];
  if (lane == 16)
    for (int v = 0; v < 8; ++v) Sp[half * MBLK + rb * 16 + 8 + v] = sacc[v];
  __syncthreads();

  // ---- epilogue: normalize, moments, qn * std + mean ----
  {
    int m0 = rb * 16;
    float rinv[8];
#pragma unroll
    for (int v = 0; v < 8; ++v) {
      int r = m0 + v + hi8;
      rinv[v] = 1.f / (Sp[r] + Sp[MBLK + r]);
    }
#pragma unroll
    for (int nt = 0; nt < 8; ++nt) {
      int c = half * 128 + nt * 16 + (lane & 15);
      float mc = mq[c], rc = rq[c];
#pragma unroll
      for (int v = 0; v < 8; ++v) {
        float mean = accM[nt][v] * rinv[v];
        float e2   = accE[nt][v] * rinv[v];
        float var  = e2 - mean * mean;
        if (var < 0.f) var = 0.f;            // relu
        float stdv = sqrtf(var);
        int row = n0g + m0 + v + hi8;
        float qn = (q[(size_t)row * 256 + c] - mc) * rc;
        out[(size_t)row * 256 + c] = qn * stdv + mean;
      }
    }
  }
}

// ---------------------------------------------------------------------------
extern "C" void kernel_launch(void* const* d_in, const int* in_sizes, int n_in,
                              void* d_out, int out_size, void* d_ws, size_t ws_size,
                              hipStream_t stream) {
  (void)n_in; (void)out_size; (void)ws_size;
  const float* q  = (const float*)d_in[0];
  const float* k  = (const float*)d_in[1];
  const float* qw = (const float*)d_in[2];
  const float* qb = (const float*)d_in[3];
  const float* kw = (const float*)d_in[4];
  const float* kb = (const float*)d_in[5];
  const float* sw = (const float*)d_in[6];
  const float* sb = (const float*)d_in[7];
  float* out = (float*)d_out;

  const int C  = in_sizes[3];       // 256
  const int N  = in_sizes[0] / C;   // 8192
  const int HW = in_sizes[1] / C;   // 9216

  char* ws = (char*)d_ws;
  size_t off = 0;
  auto alloc = [&](size_t bytes) {
    char* p = ws + off;
    off = (off + bytes + 255) & ~(size_t)255;
    return p;
  };
  float* meanq = (float*)alloc((size_t)C * 4);
  float* rstdq = (float*)alloc((size_t)C * 4);
  float* meank = (float*)alloc((size_t)C * 4);
  float* rstdk = (float*)alloc((size_t)C * 4);
  float* qbf   = (float*)alloc((size_t)C * 4);
  float* kbf   = (float*)alloc((size_t)C * 4);
  float* sbf   = (float*)alloc((size_t)C * 4);
  bf16* qwf = (bf16*)alloc((size_t)C * C * 2);
  bf16* kwf = (bf16*)alloc((size_t)C * C * 2);
  bf16* swf = (bf16*)alloc((size_t)C * C * 2);
  bf16* qe  = (bf16*)alloc((size_t)N * C * 2);   // (N, C)  A-source
  bf16* keT = (bf16*)alloc((size_t)HW * C * 2);  // (HW, C) B-source (logits)
  bf16* se  = (bf16*)alloc((size_t)C * HW * 2);  // (C, HW) B-source (mean)
  bf16* se2 = (bf16*)alloc((size_t)C * HW * 2);  // (C, HW) B-source (E[x^2])

  float qscale = rsqrtf((float)C);

  stats_kernel<<<C, 256, 0, stream>>>(q, meanq, rstdq, C, N, 1);
  stats_kernel<<<C, 256, 0, stream>>>(k, meank, rstdk, C, HW, 0);
  fold_kernel<<<C, 256, 0, stream>>>(qw, qb, kw, kb, sw, sb, meanq, rstdq, meank,
                                     rstdk, qwf, qbf, kwf, kbf, swf, sbf, C, qscale);
  linear_kernel<<<N / LNB, 256, 0, stream>>>(qwf, qbf, q, qe, nullptr, N, 0);
  linear_kernel<<<HW / LNB, 256, 0, stream>>>(kwf, kbf, k, keT, nullptr, HW, 1);
  linear_kernel<<<HW / LNB, 256, 0, stream>>>(swf, sbf, k, se, se2, HW, 2);

  // LDS: qe(32K) + 2x ke(32K) + 2x se(32K) + 2x se2(32K) + p(8K) + Sp(512B)
  size_t smem = (size_t)(MBLK * 256 + 2 * HWT * 256 + 4 * 256 * HWT + MBLK * HWT) * 2 +
                2 * MBLK * 4;
  flash_kernel<<<N / MBLK, 256, smem, stream>>>(qe, keT, se, se2, q, meanq, rstdq,
                                                out, HW);
}
